// AlignModel_89979564851327
// MI455X (gfx1250) — matrix operand order repaired and checked by reading, not compiled, not run on hardware
//
#include <hip/hip_runtime.h>
#include <hip/hip_bf16.h>
#include <stdint.h>

typedef __attribute__((ext_vector_type(16))) _Float16 v16h;
typedef __attribute__((ext_vector_type(8)))  float    v8f;

static inline int cdiv(int a, int b) { return (a + b - 1) / b; }

// ---------------------------------------------------------------------------
// device helpers
// ---------------------------------------------------------------------------
__device__ __forceinline__ float bsample1(const float* __restrict__ f, int H, int W,
                                          float y, float x) {
    y = fminf(fmaxf(y, 0.f), (float)(H - 1));
    x = fminf(fmaxf(x, 0.f), (float)(W - 1));
    float y0 = floorf(y), x0 = floorf(x);
    float wy = y - y0, wx = x - x0;
    int y0i = (int)y0, x0i = (int)x0;
    int y1i = min(y0i + 1, H - 1), x1i = min(x0i + 1, W - 1);
    float a = f[y0i * W + x0i];
    float b = f[y0i * W + x1i];
    float c = f[y1i * W + x0i];
    float d = f[y1i * W + x1i];
    return a * (1.f - wy) * (1.f - wx) + b * (1.f - wy) * wx +
           c * wy * (1.f - wx) + d * wy * wx;
}

// ---------------------------------------------------------------------------
// elementwise / data-movement kernels
// ---------------------------------------------------------------------------
__global__ void k_norm(const float* __restrict__ vid, float* __restrict__ nv,
                       int HW, int n) {
    int i = blockIdx.x * blockDim.x + threadIdx.x;
    if (i >= n) return;
    int c = (i / HW) % 3;
    float m = (c == 0) ? 0.485f : (c == 1) ? 0.456f : 0.406f;
    float s = (c == 0) ? 0.229f : (c == 1) ? 0.224f : 0.225f;
    nv[i] = (vid[i] - m) / s;
}

__global__ void k_pool2(const float* __restrict__ in, float* __restrict__ out,
                        int h, int w, int n) {   // out (NC,h,w), in (NC,2h,2w)
    int i = blockIdx.x * blockDim.x + threadIdx.x;
    if (i >= n) return;
    int x = i % w, y = (i / w) % h, q = i / (w * h);
    const float* p = in + ((size_t)q * (2 * h) + 2 * y) * (2 * w) + 2 * x;
    out[i] = 0.25f * (p[0] + p[1] + p[2 * w] + p[2 * w + 1]);
}

__global__ void k_zero(float* __restrict__ p, int n) {
    int i = blockIdx.x * blockDim.x + threadIdx.x;
    if (i < n) p[i] = 0.f;
}

__global__ void k_upflow2(const float* __restrict__ in, float* __restrict__ out,
                          int h, int w, int n) {  // in (NC,h,w) -> out (NC,2h,2w), *2
    int i = blockIdx.x * blockDim.x + threadIdx.x;
    if (i >= n) return;
    int W2 = 2 * w, H2 = 2 * h;
    int x = i % W2, y = (i / W2) % H2, q = i / (W2 * H2);
    float sy = (y + 0.5f) * 0.5f - 0.5f;
    float sx = (x + 0.5f) * 0.5f - 0.5f;
    out[i] = 2.f * bsample1(in + (size_t)q * h * w, h, w, sy, sx);
}

__global__ void k_warp3(const float* __restrict__ sup, const float* __restrict__ flow,
                        float* __restrict__ out, int H, int W, int n) { // (3img,3ch,H,W)
    int i = blockIdx.x * blockDim.x + threadIdx.x;
    if (i >= n) return;
    int HW = H * W;
    int x = i % W, y = (i / W) % H, c = (i / HW) % 3, img = i / (HW * 3);
    const float* fl = flow + (size_t)img * 2 * HW;
    float fx = fl[y * W + x];
    float fy = fl[HW + y * W + x];
    out[i] = bsample1(sup + ((size_t)img * 3 + c) * HW, H, W, (float)y + fy, (float)x + fx);
}

__global__ void k_concat8(const float* __restrict__ ref, const float* __restrict__ wrp,
                          const float* __restrict__ flow, float* __restrict__ out,
                          int H, int W, int n) {  // out (3img,8,H,W)
    int i = blockIdx.x * blockDim.x + threadIdx.x;
    if (i >= n) return;
    int HW = H * W;
    int sp = i % HW, c = (i / HW) % 8, img = i / (HW * 8);
    float v;
    if (c < 3)       v = ref [((size_t)img * 3 + c) * HW + sp];
    else if (c < 6)  v = wrp [((size_t)img * 3 + (c - 3)) * HW + sp];
    else             v = flow[((size_t)img * 2 + (c - 6)) * HW + sp];
    out[i] = v;
}

__global__ void k_add(float* __restrict__ a, const float* __restrict__ b, int n) {
    int i = blockIdx.x * blockDim.x + threadIdx.x;
    if (i < n) a[i] += b[i];
}

// ---------------------------------------------------------------------------
// weight packing: OIHW f32 -> WMMA 16x32 f16 A-fragment order
// layout: [mt][kb][lane(32)][16 halves]
// lane<16 : j0..15 -> K=kb*32+j
// lane>=16: j0..15 -> K=kb*32+16+j
// ---------------------------------------------------------------------------
__global__ void k_pack_w(const float* __restrict__ w, _Float16* __restrict__ wp,
                         int Cin, int Cout) {
    int Ktot = Cin * 49;
    int nkb = (Ktot + 31) / 32;
    int mtiles = (Cout + 15) / 16;
    int total = mtiles * nkb * 512;
    for (int e = blockIdx.x * blockDim.x + threadIdx.x; e < total;
         e += gridDim.x * blockDim.x) {
        int j    = e & 15;
        int lane = (e >> 4) & 31;
        int rest = e >> 9;
        int kb   = rest % nkb;
        int mt   = rest / nkb;
        int m    = mt * 16 + (lane & 15);
        int hi   = (lane >= 16) ? 16 : 0;
        int k    = kb * 32 + hi + j;
        float v = 0.f;
        if (m < Cout && k < Ktot) {
            int ci = k / 49;
            int r  = k - ci * 49;
            v = w[((size_t)m * Cin + ci) * 49 + r];
        }
        wp[e] = (_Float16)v;
    }
}

// ---------------------------------------------------------------------------
// 7x7 conv (pad 3) as implicit GEMM on v_wmma_f32_16x16x32_f16.
// Workgroup = 256 threads (8 waves) handles an 8x8 spatial tile (64 px = 4 N-tiles).
// Per 64-wide K chunk: cooperative im2col (64 px x 64 K f16, 8KB) into LDS, then
// each wave issues two WMMA steps for each of its two (mt,nt) 16x16 output tiles.
// WMMAs run unconditionally (EXEC all-1s, no divergent phi copies); inactive
// tiles use a clamped weight index and are simply never stored.
// ---------------------------------------------------------------------------
__global__ void k_conv7_wmma(const float* __restrict__ in, float* __restrict__ out,
                             const _Float16* __restrict__ wpack,
                             const float* __restrict__ bias,
                             int Cin, int Cout, int H, int W, int relu) {
    __shared__ __align__(64) _Float16 sB[64 * 64];

    int tilesX = (W + 7) / 8;
    int tx0 = (blockIdx.x % tilesX) * 8;
    int ty0 = (blockIdx.x / tilesX) * 8;
    int img = blockIdx.z;
    const float* inImg = in + (size_t)img * Cin * H * W;

    int lane = threadIdx.x & 31;
    int wave = threadIdx.x >> 5;

    int Ktot   = Cin * 49;
    int nkb    = (Ktot + 31) / 32;       // 32-wide WMMA steps
    int nchunk = (Ktot + 63) / 64;       // 64-wide LDS chunks
    int mtiles = (Cout + 15) / 16;

    int t0 = wave,     t1 = wave + 8;
    int nt0 = t0 & 3,  mt0 = t0 >> 2;
    int nt1 = t1 & 3,  mt1 = t1 >> 2;
    bool act0 = (mt0 < mtiles), act1 = (mt1 < mtiles);
    int mt0c = min(mt0, mtiles - 1);
    int mt1c = min(mt1, mtiles - 1);

    const _Float16* wp0 = wpack + (size_t)mt0c * nkb * 512;
    const _Float16* wp1 = wpack + (size_t)mt1c * nkb * 512;
    int bb0 = (nt0 * 16 + (lane & 15)) * 64 + ((lane >> 4) << 4);
    int bb1 = (nt1 * 16 + (lane & 15)) * 64 + ((lane >> 4) << 4);

    v8f acc0 = {};
    v8f acc1 = {};

    for (int c = 0; c < nchunk; ++c) {
        __syncthreads();   // previous chunk fully consumed
        // cooperative im2col: 64 pixels x 64 K values (zero-padded at edges/K end)
        for (int e = threadIdx.x; e < 64 * 64; e += 256) {
            int p  = e >> 6;
            int kj = e & 63;
            int k  = c * 64 + kj;
            float v = 0.f;
            if (k < Ktot) {
                int ci = k / 49;
                int r  = k - ci * 49;
                int kh = r / 7, kw = r - kh * 7;
                int y = ty0 + (p >> 3) + kh - 3;
                int x = tx0 + (p & 7)  + kw - 3;
                if (y >= 0 && y < H && x >= 0 && x < W)
                    v = inImg[(size_t)ci * H * W + y * W + x];
            }
            sB[e] = (_Float16)v;
        }
        __syncthreads();

        if (c + 1 < nchunk)   // warm next A block (global_prefetch_b8)
            __builtin_prefetch(wp0 + ((size_t)(2 * c + 2) * 32 + lane) * 16, 0, 1);

#pragma unroll
        for (int h = 0; h < 2; ++h) {
            int kb = 2 * c + h;
            if (kb >= nkb) break;   // uniform across workgroup
            v16h a0 = *(const v16h*)(wp0 + ((size_t)kb * 32 + lane) * 16);
            v16h b0 = *(const v16h*)(sB + bb0 + h * 32);
            v16h a1 = *(const v16h*)(wp1 + ((size_t)kb * 32 + lane) * 16);
            v16h b1 = *(const v16h*)(sB + bb1 + h * 32);
            acc0 = __builtin_amdgcn_wmma_f32_16x16x32_f16(
                false, a0, false, b0, (short)0, acc0, false, false);
            acc1 = __builtin_amdgcn_wmma_f32_16x16x32_f16(
                false, a1, false, b1, (short)0, acc1, false, false);
        }
    }

    float* outImg = out + (size_t)img * Cout * H * W;
    int chHi = (lane >= 16) ? 8 : 0;

    if (act0) {
        int p = nt0 * 16 + (lane & 15);
        int y = ty0 + (p >> 3), x = tx0 + (p & 7);
        if (y < H && x < W) {
            int chBase = mt0 * 16 + chHi;
#pragma unroll
            for (int r = 0; r < 8; ++r) {
                int ch = chBase + r;
                if (ch < Cout) {
                    float v = acc0[r] + bias[ch];
                    if (relu) v = fmaxf(v, 0.f);
                    outImg[(size_t)ch * H * W + y * W + x] = v;
                }
            }
        }
    }
    if (act1) {
        int p = nt1 * 16 + (lane & 15);
        int y = ty0 + (p >> 3), x = tx0 + (p & 7);
        if (y < H && x < W) {
            int chBase = mt1 * 16 + chHi;
#pragma unroll
            for (int r = 0; r < 8; ++r) {
                int ch = chBase + r;
                if (ch < Cout) {
                    float v = acc1[r] + bias[ch];
                    if (relu) v = fmaxf(v, 0.f);
                    outImg[(size_t)ch * H * W + y * W + x] = v;
                }
            }
        }
    }
}

// ---------------------------------------------------------------------------
// patch match + top-10 (one thread per (t, qy, qx, dt-block))
// ---------------------------------------------------------------------------
__global__ void k_match_topk(const float* __restrict__ vid,
                             const float* __restrict__ fflow,
                             const float* __restrict__ bflow,
                             float* __restrict__ out,
                             int T, int H, int W, int nH, int nW) {
    int gid = blockIdx.x * blockDim.x + threadIdx.x;
    int total = T * nH * nW * 3;
    if (gid >= total) return;
    int dtb = gid % 3;
    int qx  = (gid / 3) % nW;
    int qy  = (gid / 3 / nW) % nH;
    int t   = gid / 3 / nW / nH;
    int dt  = (dtb == 0) ? 0 : ((dtb == 1) ? 1 : -1);
    int HW = H * W;
    int qyi = qy * 4, qxi = qx * 4;

    float fh = 0.f, fw = 0.f;
    if (dtb == 1 && t < T - 1) {
        const float* fl = fflow + (size_t)t * 2 * HW;
        fw = fl[qyi * W + qxi];
        fh = fl[HW + qyi * W + qxi];
    } else if (dtb == 2 && t >= 1) {
        const float* fl = bflow + (size_t)(t - 1) * 2 * HW;
        fw = fl[qyi * W + qxi];
        fh = fl[HW + qyi * W + qxi];
    }
    int tid = min(max(t + dt, 0), T - 1);

    float qp[3][3][3];
    for (int c = 0; c < 3; ++c)
        for (int py = 0; py < 3; ++py) {
            int yy = min(max(qyi + py - 1, 0), H - 1);
            for (int px = 0; px < 3; ++px) {
                int xx = min(max(qxi + px - 1, 0), W - 1);
                qp[c][py][px] = vid[(((size_t)t * 3 + c) * H + yy) * W + xx];
            }
        }

    float bestd[10];
    int   besti[10];
    for (int k = 0; k < 10; ++k) { bestd[k] = 3.4e38f; besti[k] = 0; }

    for (int o = 0; o < 81; ++o) {
        int oy = o / 9 - 4, ox = o % 9 - 4;
        float d = 0.f;
        for (int py = 0; py < 3; ++py)
            for (int px = 0; px < 3; ++px) {
                float ry = (float)qyi + fh + (float)oy + (float)(py - 1);
                float rx = (float)qxi + fw + (float)ox + (float)(px - 1);
                for (int c = 0; c < 3; ++c) {
                    float cv = bsample1(vid + (((size_t)tid * 3 + c) * HW), H, W, ry, rx);
                    float df = qp[c][py][px] - cv;
                    d += df * df;
                }
            }
        int pos = 10;
        while (pos > 0 && d < bestd[pos - 1]) pos--;
        if (pos < 10) {
            for (int k = 9; k > pos; --k) { bestd[k] = bestd[k - 1]; besti[k] = besti[k - 1]; }
            bestd[pos] = d; besti[pos] = o;
        }
    }

    float* op = out + ((((size_t)t * nH + qy) * nW + qx) * 30 + dtb * 10) * 3;
    for (int k = 0; k < 10; ++k) {
        int o = besti[k];
        op[k * 3 + 0] = (float)dt;
        op[k * 3 + 1] = fh + (float)(o / 9 - 4);
        op[k * 3 + 2] = fw + (float)(o % 9 - 4);
    }
}

// ---------------------------------------------------------------------------
// host orchestration
// ---------------------------------------------------------------------------
extern "C" void kernel_launch(void* const* d_in, const int* in_sizes, int n_in,
                              void* d_out, int out_size, void* d_ws, size_t ws_size,
                              hipStream_t stream) {
    (void)in_sizes; (void)n_in; (void)out_size; (void)ws_size;
    const int T = 4, H = 192, W = 192, nH = 48, nW = 48;
    const float* vid = (const float*)d_in[0];
    const float* wArr[5]  = { (const float*)d_in[1], (const float*)d_in[3],
                              (const float*)d_in[5], (const float*)d_in[7],
                              (const float*)d_in[9] };
    const float* bArr[5]  = { (const float*)d_in[2], (const float*)d_in[4],
                              (const float*)d_in[6], (const float*)d_in[8],
                              (const float*)d_in[10] };
    const int Cins[5]  = { 8, 32, 64, 32, 16 };
    const int Couts[5] = { 32, 64, 32, 16, 2 };

    uint8_t* wsb = (uint8_t*)d_ws;
    size_t off = 0;
    auto take = [&](size_t bytes) {
        size_t o = off;
        off += (bytes + 255) & ~(size_t)255;
        return o;
    };

    size_t o_nv = take((size_t)T * 3 * H * W * 4);
    size_t o_p1 = take((size_t)T * 3 * 96 * 96 * 4);
    size_t o_p2 = take((size_t)T * 3 * 48 * 48 * 4);
    size_t o_p3 = take((size_t)T * 3 * 24 * 24 * 4);
    size_t o_wp[4][5];
    for (int l = 0; l < 4; ++l)
        for (int j = 0; j < 5; ++j) {
            int nkb = cdiv(Cins[j] * 49, 32);
            int mtiles = cdiv(Couts[j], 16);
            o_wp[l][j] = take((size_t)mtiles * nkb * 512 * 2);
        }
    size_t o_fA = take((size_t)3 * 2 * H * W * 4);
    size_t o_fB = take((size_t)3 * 2 * H * W * 4);
    size_t o_wr = take((size_t)3 * 3 * H * W * 4);
    size_t o_bA = take((size_t)3 * 64 * H * W * 4);
    size_t o_bB = take((size_t)3 * 32 * H * W * 4);
    size_t o_ff = take((size_t)3 * 2 * H * W * 4);
    size_t o_bf = take((size_t)3 * 2 * H * W * 4);

    float* nv  = (float*)(wsb + o_nv);
    float* p1  = (float*)(wsb + o_p1);
    float* p2  = (float*)(wsb + o_p2);
    float* p3  = (float*)(wsb + o_p3);
    float* fA  = (float*)(wsb + o_fA);
    float* fB  = (float*)(wsb + o_fB);
    float* wrp = (float*)(wsb + o_wr);
    float* bA  = (float*)(wsb + o_bA);
    float* bB  = (float*)(wsb + o_bB);
    float* ff  = (float*)(wsb + o_ff);
    float* bf  = (float*)(wsb + o_bf);

    const int BS = 256;
    auto g1 = [&](int n) { return dim3((unsigned)cdiv(n, BS)); };

    // normalize + pyramids
    int nElems = T * 3 * H * W;
    k_norm<<<g1(nElems), BS, 0, stream>>>(vid, nv, H * W, nElems);
    k_pool2<<<g1(T * 3 * 96 * 96), BS, 0, stream>>>(nv, p1, 96, 96, T * 3 * 96 * 96);
    k_pool2<<<g1(T * 3 * 48 * 48), BS, 0, stream>>>(p1, p2, 48, 48, T * 3 * 48 * 48);
    k_pool2<<<g1(T * 3 * 24 * 24), BS, 0, stream>>>(p2, p3, 24, 24, T * 3 * 24 * 24);

    // pack weights (per level, per layer) into WMMA A-fragment order
    for (int l = 0; l < 4; ++l)
        for (int j = 0; j < 5; ++j) {
            int nkb = cdiv(Cins[j] * 49, 32);
            int mtiles = cdiv(Couts[j], 16);
            int total = mtiles * nkb * 512;
            k_pack_w<<<g1(total), BS, 0, stream>>>(
                wArr[j] + (size_t)l * Couts[j] * Cins[j] * 49,
                (_Float16*)(wsb + o_wp[l][j]), Cins[j], Couts[j]);
        }

    float* pyr[4] = { nv, p1, p2, p3 };
    const int Hs[4] = { 192, 96, 48, 24 };

    auto conv = [&](const float* in, float* out, size_t wpoff, const float* bias,
                    int Cin, int Cout, int Hl, int Wl, int relu) {
        dim3 g((unsigned)(cdiv(Wl, 8) * cdiv(Hl, 8)), 1, 3);
        k_conv7_wmma<<<g, BS, 0, stream>>>(in, out, (const _Float16*)(wsb + wpoff),
                                           bias, Cin, Cout, Hl, Wl, relu);
    };

    for (int dir = 0; dir < 2; ++dir) {
        float* cur = fA;
        float* oth = fB;
        k_zero<<<g1(3 * 2 * 24 * 24), BS, 0, stream>>>(cur, 3 * 2 * 24 * 24);
        for (int l = 3; l >= 0; --l) {
            int Hl = Hs[l], Wl = Hs[l];
            if (l < 3) {
                int n = 3 * 2 * Hl * Wl;
                k_upflow2<<<g1(n), BS, 0, stream>>>(cur, oth, Hl / 2, Wl / 2, n);
                float* t = cur; cur = oth; oth = t;
            }
            size_t frame = (size_t)3 * Hl * Wl;
            const float* refp = pyr[l] + (dir == 0 ? 0 : 1) * frame;
            const float* supp = pyr[l] + (dir == 0 ? 1 : 0) * frame;

            int nw = 3 * 3 * Hl * Wl;
            k_warp3<<<g1(nw), BS, 0, stream>>>(supp, cur, wrp, Hl, Wl, nw);
            int nc = 3 * 8 * Hl * Wl;
            k_concat8<<<g1(nc), BS, 0, stream>>>(refp, wrp, cur, bA, Hl, Wl, nc);

            conv(bA, bB, o_wp[l][0], bArr[0] + (size_t)l * 32, 8,  32, Hl, Wl, 1);
            conv(bB, bA, o_wp[l][1], bArr[1] + (size_t)l * 64, 32, 64, Hl, Wl, 1);
            conv(bA, bB, o_wp[l][2], bArr[2] + (size_t)l * 32, 64, 32, Hl, Wl, 1);
            conv(bB, bA, o_wp[l][3], bArr[3] + (size_t)l * 16, 32, 16, Hl, Wl, 1);
            conv(bA, bB, o_wp[l][4], bArr[4] + (size_t)l * 2,  16, 2,  Hl, Wl, 0);

            int na = 3 * 2 * Hl * Wl;
            k_add<<<g1(na), BS, 0, stream>>>(cur, bB, na);
        }
        hipMemcpyAsync(dir == 0 ? ff : bf, cur, (size_t)3 * 2 * H * W * 4,
                       hipMemcpyDeviceToDevice, stream);
    }

    int nm = T * nH * nW * 3;
    k_match_topk<<<g1(nm), BS, 0, stream>>>(vid, ff, bf, (float*)d_out,
                                            T, H, W, nH, nW);
}
